// IndependentMLPsVariableOutputPredictor_71734543778319
// MI455X (gfx1250) — matrix operand order repaired and checked by reading, not compiled, not run on hardware
//
#include <hip/hip_runtime.h>
#include <math.h>

typedef _Float16 h16;
typedef __attribute__((ext_vector_type(16))) _Float16 v16h;
typedef __attribute__((ext_vector_type(8)))  _Float16 v8h;
typedef __attribute__((ext_vector_type(4)))  _Float16 v4h;
typedef __attribute__((ext_vector_type(8)))  float    v8f;
typedef __attribute__((ext_vector_type(4)))  float    v4f;
typedef unsigned int u32;
typedef __attribute__((ext_vector_type(4))) u32 u32x4;
typedef __attribute__((ext_vector_type(4))) int i32x4;
typedef __attribute__((ext_vector_type(8))) int i32x8;

#define NPART 16
#define DLAT  768
#define DHID  3072
#define NB    256
#define MAXO  250
#define OUTW  2800

#define BN       64          // weight rows (N columns of C) staged per block
#define KC       128         // K-chunk staged per iteration
#define NTHREADS 256         // 8 wave32s
#define SMEM_F32_BYTES (BN * KC * 4)
#define SMEM_F16_BYTES (BN * KC * 2)

#if defined(__has_builtin)
#if __has_builtin(__builtin_amdgcn_tensor_load_to_lds) && \
    __has_builtin(__builtin_amdgcn_s_wait_tensorcnt)
#define HAVE_TDM 1
#endif
#endif
#ifndef HAVE_TDM
#define HAVE_TDM 0
#endif

// ---------------------------------------------------------------------------
// Kernel 1: per-(b,p) LayerNorm over D=768, emit f16 normalized activations.
// ---------------------------------------------------------------------------
__global__ __launch_bounds__(256) void ln_f16_kernel(
    const float* __restrict__ x, const float* __restrict__ gamma,
    const float* __restrict__ beta, h16* __restrict__ xn) {
  const int bp = blockIdx.x;          // b*NPART + p
  const int p  = bp & (NPART - 1);
  const float* xr = x + (size_t)bp * DLAT;

  __shared__ float s_sum[256];
  __shared__ float s_sq[256];
  float lsum = 0.f, lsq = 0.f;
  for (int i = threadIdx.x; i < DLAT; i += 256) {
    float v = xr[i];
    lsum += v;
    lsq  += v * v;
  }
  s_sum[threadIdx.x] = lsum;
  s_sq[threadIdx.x]  = lsq;
  __syncthreads();
  for (int off = 128; off > 0; off >>= 1) {
    if (threadIdx.x < off) {
      s_sum[threadIdx.x] += s_sum[threadIdx.x + off];
      s_sq[threadIdx.x]  += s_sq[threadIdx.x + off];
    }
    __syncthreads();
  }
  const float mu   = s_sum[0] * (1.0f / DLAT);
  const float var  = s_sq[0] * (1.0f / DLAT) - mu * mu;
  const float rstd = rsqrtf(var + 1e-5f);

  const float* g  = gamma + (size_t)p * DLAT;
  const float* bt = beta  + (size_t)p * DLAT;
  h16* xo = xn + (size_t)bp * DLAT;
  for (int i = threadIdx.x; i < DLAT; i += 256) {
    float v = (xr[i] - mu) * rstd * g[i] + bt[i];
    xo[i] = (h16)v;
  }
}

// ---------------------------------------------------------------------------
// WMMA fragment helpers (wave32 layouts per CDNA5 ISA 7.12.2).
// ---------------------------------------------------------------------------
__device__ __forceinline__ v16h load_a_frag_f16(const h16* rowK, int grp) {
  v8h a0 = *(const v8h*)(rowK + grp * 8);
  v8h a1 = *(const v8h*)(rowK + 16 + grp * 8);
  v16h A;
#pragma unroll
  for (int i = 0; i < 8; ++i) { A[i] = a0[i]; A[i + 8] = a1[i]; }
  return A;
}

__device__ __forceinline__ v16h load_b_frag_lds(const h16* rowK, int grp) {
  const h16* q = rowK + grp * 8;         // K [8g,8g+8) and [16+8g,24+8g) per lane
  v8h b0 = *(const v8h*)(q);
  v8h b1 = *(const v8h*)(q + 16);
  v16h Bv;
#pragma unroll
  for (int i = 0; i < 8; ++i) { Bv[i] = b0[i]; Bv[i + 8] = b1[i]; }
  return Bv;
}

// ---------------------------------------------------------------------------
// Stage a BN x KC f32 weight tile into LDS.
// TDM path: one wave issues TENSOR_LOAD_TO_LDS with a 2D D# (ISA ch.8):
//   group0: count=1, lds_addr, global_addr[56:0], type=2
//   group1: data_size=2(4B), tensor_dim0=k_rem, tensor_dim1=rows_rem (row
//           OOB -> zero fill, which clamps padded w2 rows), tile=KCxBN,
//           tensor_dim0_stride=ld
// Fallback: cooperative vector loads with explicit row clamp.
// ---------------------------------------------------------------------------
__device__ __forceinline__ void stage_weights(const float* __restrict__ src,
                                              float* sB32, u32 lds_byte_off,
                                              int rows_rem, int ld, int k_rem,
                                              int tid) {
#if HAVE_TDM
  if (tid < 32) {
    unsigned long long ga = (unsigned long long)(uintptr_t)src;
    u32x4 g0;
    g0[0] = 1u;                                            // count=1
    g0[1] = lds_byte_off;                                  // lds_addr
    g0[2] = (u32)(ga & 0xffffffffu);                       // global_addr lo
    g0[3] = (u32)((ga >> 32) & 0x01ffffffu) | (2u << 30);  // addr hi + type=2
    i32x8 g1;
    g1[0] = (2 << 16);                                     // data_size = 4B
    g1[1] = (int)(((u32)k_rem & 0xffffu) << 16);           // tensor_dim0 lo16
    g1[2] = (int)((((u32)k_rem >> 16) & 0xffffu) |
                  (((u32)rows_rem & 0xffffu) << 16));      // td0 hi | td1 lo
    g1[3] = (int)(((((u32)rows_rem) >> 16) & 0xffffu) |
                  (((u32)KC & 0xffffu) << 16));            // td1 hi | tile_dim0
    g1[4] = (int)((u32)BN & 0xffffu);                      // tile_dim1 (tile_dim2=0)
    g1[5] = ld;                                            // tensor_dim0_stride lo32
    g1[6] = 0;
    g1[7] = 0;
    i32x4 z4 = {0, 0, 0, 0};
    i32x8 z8 = {0, 0, 0, 0, 0, 0, 0, 0};
    __builtin_amdgcn_tensor_load_to_lds(g0, g1, z4, z4, z8, 0);
    __builtin_amdgcn_s_wait_tensorcnt((short)0);
  }
  (void)sB32;
#else
  (void)lds_byte_off;
  for (int i = tid; i < BN * KC; i += NTHREADS) {
    const int r = i / KC;
    const int k = i - r * KC;
    sB32[i] = (r < rows_rem) ? src[(size_t)r * ld + k] : 0.0f;
  }
  (void)k_rem;
#endif
}

__device__ __forceinline__ void convert_tile_f16(const float* sB32, h16* sB16,
                                                 int tid) {
  const v4f* src = (const v4f*)sB32;
#pragma unroll
  for (int j = 0; j < (BN * KC / 4) / NTHREADS; ++j) {  // 8 iterations
    const int idx = tid + j * NTHREADS;
    v4f v = src[idx];
    v4h o;
#pragma unroll
    for (int i = 0; i < 4; ++i) o[i] = (h16)v[i];
    *(v4h*)(sB16 + idx * 4) = o;
  }
}

// ---------------------------------------------------------------------------
// Kernel 2: grouped GEMM1 + exact GELU.
// Block tile: 128(M) x 64(N); wave w -> M rows [16w,16w+16), all 4 N-tiles.
// One A-fragment feeds 4 WMMAs; weight tile staged once via TDM + LDS.
// grid = (DHID/BN, NB/128, NPART)
// ---------------------------------------------------------------------------
__global__ __launch_bounds__(NTHREADS) void gemm1_gelu_kernel(
    const h16* __restrict__ xn, const float* __restrict__ w1,
    h16* __restrict__ hbuf) {
  __shared__ __align__(16) unsigned char smem[SMEM_F32_BYTES + SMEM_F16_BYTES];
  float* sB32 = (float*)smem;                     // LDS offset 0 (sole object)
  h16*   sB16 = (h16*)(smem + SMEM_F32_BYTES);

  const int n0    = blockIdx.x * BN;
  const int p     = blockIdx.z;
  const int tid   = threadIdx.x;
  const int wave  = tid >> 5;
  const int lane  = tid & 31;
  const int laneM = lane & 15;
  const int grp   = lane >> 4;
  const int m0    = blockIdx.y * 128 + wave * 16;

  const h16*   aRow  = xn + (((size_t)(m0 + laneM)) * NPART + p) * DLAT;
  const float* wbase = w1 + ((size_t)p * DHID + n0) * DLAT;

  v8f acc[4] = {};
  for (int c = 0; c < DLAT / KC; ++c) {
    __syncthreads();
    stage_weights(wbase + (size_t)c * KC, sB32, /*lds_off=*/0u,
                  DHID - n0, DLAT, DLAT - c * KC, tid);
    __syncthreads();
    convert_tile_f16(sB32, sB16, tid);
    __syncthreads();

    const int kg = c * KC;
    __builtin_prefetch(aRow + kg + KC, 0, 1);     // next A slab (speculative)
#pragma unroll
    for (int ks = 0; ks < KC; ks += 32) {
      v16h A = load_a_frag_f16(aRow + kg + ks, grp);
#pragma unroll
      for (int t = 0; t < 4; ++t) {
        v16h Bv = load_b_frag_lds(sB16 + (t * 16 + laneM) * KC + ks, grp);
        acc[t] = __builtin_amdgcn_wmma_f32_16x16x32_f16(
            false, A, false, Bv, (short)0, acc[t], false, false);
      }
    }
  }

#pragma unroll
  for (int t = 0; t < 4; ++t) {
    const int col = n0 + t * 16 + laneM;
#pragma unroll
    for (int r = 0; r < 8; ++r) {
      const int m = m0 + r + grp * 8;
      float v = acc[t][r];
      v = 0.5f * v * (1.0f + erff(v * 0.70710678118654752f));  // exact GELU
      hbuf[(((size_t)m) * NPART + p) * DHID + col] = (h16)v;
    }
  }
}

// ---------------------------------------------------------------------------
// Kernel 3: grouped GEMM2 + bias + variable-width scatter.
// Same tiling; padded w2 rows are zero-filled by the TDM tensor_dim1 clamp
// (or the fallback's explicit clamp); stores masked to col < out_dims[p].
// grid = (256/BN, NB/128, NPART)
// ---------------------------------------------------------------------------
__global__ __launch_bounds__(NTHREADS) void gemm2_bias_kernel(
    const h16* __restrict__ hbuf, const float* __restrict__ w2,
    const float* __restrict__ bias2, float* __restrict__ out) {
  __shared__ __align__(16) unsigned char smem[SMEM_F32_BYTES + SMEM_F16_BYTES];
  float* sB32 = (float*)smem;
  h16*   sB16 = (h16*)(smem + SMEM_F32_BYTES);

  const int n0    = blockIdx.x * BN;              // 0,64,128,192 (padded N=256)
  const int p     = blockIdx.z;
  const int tid   = threadIdx.x;
  const int wave  = tid >> 5;
  const int lane  = tid & 31;
  const int laneM = lane & 15;
  const int grp   = lane >> 4;
  const int m0    = blockIdx.y * 128 + wave * 16;
  const int rows_rem = MAXO - n0;                 // 58 for the last N-block

  const h16*   aRow  = hbuf + (((size_t)(m0 + laneM)) * NPART + p) * DHID;
  const float* wbase = w2 + ((size_t)p * MAXO + n0) * DHID;

  v8f acc[4] = {};
  for (int c = 0; c < DHID / KC; ++c) {
    __syncthreads();
    stage_weights(wbase + (size_t)c * KC, sB32, /*lds_off=*/0u,
                  rows_rem, DHID, DHID - c * KC, tid);
    __syncthreads();
    convert_tile_f16(sB32, sB16, tid);
    __syncthreads();

    const int kg = c * KC;
    __builtin_prefetch(aRow + kg + KC, 0, 1);
#pragma unroll
    for (int ks = 0; ks < KC; ks += 32) {
      v16h A = load_a_frag_f16(aRow + kg + ks, grp);
#pragma unroll
      for (int t = 0; t < 4; ++t) {
        v16h Bv = load_b_frag_lds(sB16 + (t * 16 + laneM) * KC + ks, grp);
        acc[t] = __builtin_amdgcn_wmma_f32_16x16x32_f16(
            false, A, false, Bv, (short)0, acc[t], false, false);
      }
    }
  }

  const int odim = 100 + 10 * p;
  const int ooff = 100 * p + 5 * p * (p - 1);
#pragma unroll
  for (int t = 0; t < 4; ++t) {
    const int col = n0 + t * 16 + laneM;
    if (col < odim) {
      const float bb = bias2[(size_t)p * MAXO + col];
#pragma unroll
      for (int r = 0; r < 8; ++r) {
        const int m = m0 + r + grp * 8;
        out[(size_t)m * OUTW + ooff + col] = acc[t][r] + bb;
      }
    }
  }
}

// ---------------------------------------------------------------------------
extern "C" void kernel_launch(void* const* d_in, const int* in_sizes, int n_in,
                              void* d_out, int out_size, void* d_ws, size_t ws_size,
                              hipStream_t stream) {
  const float* x     = (const float*)d_in[0];
  const float* gamma = (const float*)d_in[1];
  const float* beta  = (const float*)d_in[2];
  const float* w1    = (const float*)d_in[3];
  const float* w2    = (const float*)d_in[4];
  const float* b2    = (const float*)d_in[5];
  float* out = (float*)d_out;

  // workspace: f16 normalized activations (6.3 MB) + f16 hidden (25.2 MB)
  h16* xn   = (h16*)d_ws;
  h16* hbuf = (h16*)((char*)d_ws + (size_t)NB * NPART * DLAT * sizeof(h16));

  ln_f16_kernel<<<NB * NPART, 256, 0, stream>>>(x, gamma, beta, xn);
  gemm1_gelu_kernel<<<dim3(DHID / BN, NB / 128, NPART), NTHREADS, 0, stream>>>(
      xn, w1, hbuf);
  gemm2_bias_kernel<<<dim3(256 / BN, NB / 128, NPART), NTHREADS, 0, stream>>>(
      hbuf, w2, b2, out);
}
